// GCN_fusion7_91036126806366
// MI455X (gfx1250) — matrix operand order repaired and checked by reading, not compiled, not run on hardware
//
#include <hip/hip_runtime.h>
#include <hip/hip_bf16.h>

typedef __attribute__((ext_vector_type(16))) __bf16 v16bf;
typedef __attribute__((ext_vector_type(8)))  float  v8f;

#define NNODE 50000
#define NEDGE 800000
#define FIN   128
#define HID   128
#define CDIM  64
#define KS    144   // padded LDS row stride (halfs): 288B rows -> 32B-aligned v16bf loads

// ---------------------------------------------------------------- zero fill
__global__ __launch_bounds__(256) void zero_kernel(float* __restrict__ p, long n) {
    long i = (long)blockIdx.x * blockDim.x + threadIdx.x;
    long stride = (long)gridDim.x * blockDim.x;
    for (; i < n; i += stride) p[i] = 0.0f;
}

// ---------------------------------------------------------------- GEMM1: hw1 = x @ W1   [N,128]x[128,128]
__global__ __launch_bounds__(256) void gemm1_kernel(const float* __restrict__ x,
                                                    const float* __restrict__ W1,
                                                    float* __restrict__ hw1) {
    __shared__ __bf16 ldsB[FIN * KS];          // W1 transposed: ldsB[n*KS + k] = W1[k][n]
    const int tid = threadIdx.x;
    for (int i = tid; i < FIN * HID; i += 256) {
        int k = i >> 7, n = i & 127;
        ldsB[n * KS + k] = (__bf16)W1[i];
    }
    __syncthreads();

    const int wave = tid >> 5;
    const int lane = tid & 31;
    const int mtile = blockIdx.x * 8 + wave;         // 3125 tiles total
    if (mtile >= 3125) return;                       // whole-wave skip (EXEC all-1 for WMMA)

    const int half = lane >> 4;                      // 0: lanes 0-15, 1: lanes 16-31
    const int mrow = mtile * 16 + (lane & 15);
    const float* xrow = x + (long)mrow * FIN;

    // A operand: contiguous k order used here (same as B). lanes0-15: K 0-15 ; lanes16-31: K 16-31
    v16bf a[4];
#pragma unroll
    for (int kk = 0; kk < 4; ++kk) {
        const int c0 = kk * 32 + half * 16;
#pragma unroll
        for (int j = 0; j < 8; ++j) {
            a[kk][j]     = (__bf16)xrow[c0 + j];
            a[kk][8 + j] = (__bf16)xrow[c0 + 8 + j];
        }
    }
#pragma unroll
    for (int nt = 0; nt < 8; ++nt) {                 // 8 N-tiles of 16 -> 128 cols
        v8f acc = {};
        const int n = nt * 16 + (lane & 15);
#pragma unroll
        for (int kk = 0; kk < 4; ++kk) {
            const int koff = kk * 32 + half * 16;    // B: lanes0-15 K 0-15, lanes16-31 K 16-31
            v16bf b = *(const v16bf*)&ldsB[n * KS + koff];
            acc = __builtin_amdgcn_wmma_f32_16x16x32_bf16(
                      false, a[kk], false, b, (short)0, acc, false, false);
        }
        // D layout: vgpr i -> row (half*8 + i), col (lane&15)
        float* orow = hw1 + (long)(mtile * 16 + half * 8) * HID + nt * 16 + (lane & 15);
#pragma unroll
        for (int i = 0; i < 8; ++i) orow[(long)i * HID] = acc[i];
    }
}

// ---------------------------------------------------------------- GEMM2: hw2 = relu(agg1+b1) @ W2  [N,128]x[128,64]
__global__ __launch_bounds__(256) void gemm2_kernel(const float* __restrict__ agg1,
                                                    const float* __restrict__ b1,
                                                    const float* __restrict__ W2,
                                                    float* __restrict__ hw2) {
    __shared__ __bf16 ldsB[CDIM * KS];               // W2 transposed
    __shared__ float  ldsb1[HID];
    const int tid = threadIdx.x;
    for (int i = tid; i < HID * CDIM; i += 256) {
        int k = i >> 6, n = i & 63;
        ldsB[n * KS + k] = (__bf16)W2[i];
    }
    if (tid < HID) ldsb1[tid] = b1[tid];
    __syncthreads();

    const int wave = tid >> 5;
    const int lane = tid & 31;
    const int mtile = blockIdx.x * 8 + wave;
    if (mtile >= 3125) return;

    const int half = lane >> 4;
    const int mrow = mtile * 16 + (lane & 15);
    const float* arow = agg1 + (long)mrow * HID;

    v16bf a[4];
#pragma unroll
    for (int kk = 0; kk < 4; ++kk) {
        const int c0 = kk * 32 + half * 16;
#pragma unroll
        for (int j = 0; j < 8; ++j) {
            float v0 = arow[c0 + j]      + ldsb1[c0 + j];
            float v1 = arow[c0 + 8 + j] + ldsb1[c0 + 8 + j];
            a[kk][j]     = (__bf16)(v0 > 0.0f ? v0 : 0.0f);   // fused bias+ReLU
            a[kk][8 + j] = (__bf16)(v1 > 0.0f ? v1 : 0.0f);
        }
    }
#pragma unroll
    for (int nt = 0; nt < 4; ++nt) {                 // 64 output cols
        v8f acc = {};
        const int n = nt * 16 + (lane & 15);
#pragma unroll
        for (int kk = 0; kk < 4; ++kk) {
            const int koff = kk * 32 + half * 16;
            v16bf b = *(const v16bf*)&ldsB[n * KS + koff];
            acc = __builtin_amdgcn_wmma_f32_16x16x32_bf16(
                      false, a[kk], false, b, (short)0, acc, false, false);
        }
        float* orow = hw2 + (long)(mtile * 16 + half * 8) * CDIM + nt * 16 + (lane & 15);
#pragma unroll
        for (int i = 0; i < 8; ++i) orow[(long)i * CDIM] = acc[i];
    }
}

// ---------------------------------------------------------------- edge aggregation, 128 feats (wave per edge)
__global__ __launch_bounds__(256) void agg128_kernel(const float* __restrict__ hw,
                                                     const int* __restrict__ src,
                                                     const int* __restrict__ dst,
                                                     const float* __restrict__ ew,
                                                     float* __restrict__ agg) {
    const int wave = threadIdx.x >> 5, lane = threadIdx.x & 31;
    const long e = (long)blockIdx.x * 8 + wave;
    if (e >= NEDGE) return;
    const int  s = src[e], d = dst[e];
    const float w = ew[e];
    float4 v = *((const float4*)(hw + (long)s * 128) + lane);
    float* pd = agg + (long)d * 128 + lane * 4;
    unsafeAtomicAdd(pd + 0, w * v.x);
    unsafeAtomicAdd(pd + 1, w * v.y);
    unsafeAtomicAdd(pd + 2, w * v.z);
    unsafeAtomicAdd(pd + 3, w * v.w);
}

// ---------------------------------------------------------------- edge aggregation, 64 feats
__global__ __launch_bounds__(256) void agg64_kernel(const float* __restrict__ hw,
                                                    const int* __restrict__ src,
                                                    const int* __restrict__ dst,
                                                    const float* __restrict__ ew,
                                                    float* __restrict__ agg) {
    const int wave = threadIdx.x >> 5, lane = threadIdx.x & 31;
    const long e = (long)blockIdx.x * 8 + wave;
    if (e >= NEDGE) return;
    const int  s = src[e], d = dst[e];
    const float w = ew[e];
    float2 v = *((const float2*)(hw + (long)s * 64) + lane);
    float* pd = agg + (long)d * 64 + lane * 2;
    unsafeAtomicAdd(pd + 0, w * v.x);
    unsafeAtomicAdd(pd + 1, w * v.y);
}

// ---------------------------------------------------------------- column sums of relu(agg2 + b2)  (h2 never materialized)
__global__ __launch_bounds__(256) void colsum_kernel(const float* __restrict__ agg2,
                                                     const float* __restrict__ b2,
                                                     float* __restrict__ colsum) {
    __shared__ float red[256];
    const int tid = threadIdx.x;
    const int col = tid & 63;
    const float bias = b2[col];
    float s = 0.0f;
    const long total = (long)NNODE * CDIM;
    for (long i = (long)blockIdx.x * 256 + tid; i < total; i += (long)gridDim.x * 256) {
        float v = agg2[i] + bias;           // (i % 64) == col since strides are multiples of 64
        s += (v > 0.0f ? v : 0.0f);
    }
    red[tid] = s;
    __syncthreads();
    if (tid < 64) {
        float t = red[tid] + red[tid + 64] + red[tid + 128] + red[tid + 192];
        unsafeAtomicAdd(colsum + tid, t);
    }
}

// ---------------------------------------------------------------- tiny fusion head: selu/fc/attention/softmax -> out[64]
__global__ __launch_bounds__(64) void final_kernel(const float* __restrict__ colsum,
                                                   const float* __restrict__ sub_fea,
                                                   const float* __restrict__ fcW,
                                                   const float* __restrict__ fcb,
                                                   const float* __restrict__ attW,
                                                   const float* __restrict__ atta,
                                                   float* __restrict__ out) {
    __shared__ float tok[2][64];
    __shared__ float red[64];
    __shared__ float sc[8];     // scores [t*4+h]
    __shared__ float wgt[8];    // softmax weights [t*4+h]
    __shared__ float mx_s, lse_s;
    const int c = threadIdx.x;  // 0..63

    // token 0: g = selu(mean(h2))
    float m = colsum[c] * (1.0f / (float)NNODE);
    const float SC = 1.0507009873554805f, AL = 1.6732632423543772f;
    tok[0][c] = (m > 0.0f) ? SC * m : SC * AL * (expf(m) - 1.0f);

    // token 1: x_ext = sub_fea @ fcW + fcb
    float acc = fcb[c];
    for (int k = 0; k < 256; ++k) acc += sub_fea[k] * fcW[k * 64 + c];
    tok[1][c] = acc;
    __syncthreads();

    // scores[t,h] = sum_k tanh(sum_d tok[t][d]*attW[h,d,k]) * atta[h,k]
    for (int t = 0; t < 2; ++t)
        for (int h = 0; h < 4; ++h) {
            float e = 0.0f;
            for (int d2 = 0; d2 < 64; ++d2) e += tok[t][d2] * attW[(h * 64 + d2) * 64 + c];
            e = tanhf(e);
            red[c] = e * atta[h * 64 + c];
            __syncthreads();
            if (c == 0) {
                float ssum = 0.0f;
                for (int i = 0; i < 64; ++i) ssum += red[i];
                sc[t * 4 + h] = ssum;
            }
            __syncthreads();
        }

    // softmax over the 2 tokens, per head
    if (c < 4) {
        float s0 = sc[c], s1 = sc[4 + c];
        float mx = fmaxf(s0, s1);
        float e0 = expf(s0 - mx), e1 = expf(s1 - mx);
        float inv = 1.0f / (e0 + e1);
        wgt[c] = e0 * inv;
        wgt[4 + c] = e1 * inv;
    }
    __syncthreads();

    // fused[c] = mean_h sum_t w[t,h]*tok[t][c]
    float fused = 0.0f;
    for (int h = 0; h < 4; ++h) fused += wgt[h] * tok[0][c] + wgt[4 + h] * tok[1][c];
    fused *= 0.25f;

    // log_softmax over the 64 classes
    red[c] = fused;
    __syncthreads();
    if (c == 0) {
        float mx = red[0];
        for (int i = 1; i < 64; ++i) mx = fmaxf(mx, red[i]);
        float se = 0.0f;
        for (int i = 0; i < 64; ++i) se += expf(red[i] - mx);
        mx_s = mx;
        lse_s = logf(se);
    }
    __syncthreads();
    out[c] = fused - mx_s - lse_s;
}

// ---------------------------------------------------------------- host launcher
extern "C" void kernel_launch(void* const* d_in, const int* in_sizes, int n_in,
                              void* d_out, int out_size, void* d_ws, size_t ws_size,
                              hipStream_t stream) {
    (void)in_sizes; (void)n_in; (void)out_size; (void)ws_size;
    const float* x    = (const float*)d_in[0];
    const int*   eidx = (const int*)  d_in[1];   // [2,E]: src then dst
    const float* ew   = (const float*)d_in[2];
    const float* sub  = (const float*)d_in[3];
    const float* W1   = (const float*)d_in[4];
    const float* b1   = (const float*)d_in[5];
    const float* W2   = (const float*)d_in[6];
    const float* b2   = (const float*)d_in[7];
    const float* fcW  = (const float*)d_in[8];
    const float* fcb  = (const float*)d_in[9];
    const float* attW = (const float*)d_in[10];
    const float* atta = (const float*)d_in[11];
    float* out = (float*)d_out;

    float* ws     = (float*)d_ws;
    float* hw1    = ws;                       // N*128
    float* agg1   = hw1  + (long)NNODE * 128; // N*128
    float* hw2    = agg1 + (long)NNODE * 128; // N*64
    float* agg2   = hw2  + (long)NNODE * 64;  // N*64
    float* colsum = agg2 + (long)NNODE * 64;  // 64

    const int* esrc = eidx;
    const int* edst = eidx + NEDGE;

    zero_kernel<<<2048, 256, 0, stream>>>(agg1, (long)NNODE * 128);
    zero_kernel<<<1024, 256, 0, stream>>>(agg2, (long)NNODE * 64 + 64); // agg2 + colsum (contiguous)

    gemm1_kernel<<<391, 256, 0, stream>>>(x, W1, hw1);
    agg128_kernel<<<NEDGE / 8, 256, 0, stream>>>(hw1, esrc, edst, ew, agg1);
    gemm2_kernel<<<391, 256, 0, stream>>>(agg1, b1, W2, hw2);
    agg64_kernel<<<NEDGE / 8, 256, 0, stream>>>(hw2, esrc, edst, ew, agg2);
    colsum_kernel<<<512, 256, 0, stream>>>(agg2, b2, colsum);
    final_kernel<<<1, 64, 0, stream>>>(colsum, sub, fcW, fcb, attW, atta, out);
}
